// Decoder_12412455485600
// MI455X (gfx1250) — matrix operand order repaired and checked by reading, not compiled, not run on hardware
//
#include <hip/hip_runtime.h>
#include <stdint.h>

typedef __attribute__((ext_vector_type(16))) __bf16 v16bf;
typedef __attribute__((ext_vector_type(8)))  float  v8f;

__device__ __forceinline__ unsigned short f2bf(float f) {
  unsigned u = __float_as_uint(f);
  u += 0x7fffu + ((u >> 16) & 1u);           // round-to-nearest-even
  return (unsigned short)(u >> 16);
}
__device__ __forceinline__ float bf2f(unsigned short h) {
  return __uint_as_float(((unsigned)h) << 16);
}
__device__ __forceinline__ float sigf(float x) { return 1.f / (1.f + __expf(-x)); }

union Frag16 { v16bf v; uint4 q[2]; };

// gfx1250 async DMA: copy 16 bytes global -> LDS, tracked by ASYNCcnt.
__device__ __forceinline__ void async_copy16(unsigned lds_byte_off,
                                             const void* gbase,
                                             unsigned gbyte_off) {
  asm volatile("global_load_async_to_lds_b128 %0, %1, %2"
               :: "v"(lds_byte_off), "v"(gbyte_off), "s"(gbase)
               : "memory");
}
__device__ __forceinline__ void wait_async0() {
  asm volatile("s_wait_asynccnt 0" ::: "memory");
}

// ---------------- input conversion: (16,64,128,128) NCHW f32 -> NHWC bf16 ----
__global__ void convert_input_kernel(const float* __restrict__ src,
                                     unsigned short* __restrict__ dst) {
  const size_t HW = 128 * 128;
  const size_t total = (size_t)16 * 64 * HW;
  for (size_t t = (size_t)blockIdx.x * blockDim.x + threadIdx.x; t < total;
       t += (size_t)gridDim.x * blockDim.x) {
    size_t n  = t / (64 * HW);
    size_t r  = t - n * (64 * HW);
    size_t c  = r / HW;
    size_t yx = r - c * HW;
    dst[(n * HW + yx) * 64 + c] = f2bf(src[t]);
  }
}

// ------------- weight packing: OIHW f32 -> bf16 WMMA B-fragment layout -------
// K index = tap*64 + ic ; fragment (kc,nf): 32 lanes x 16 bf16, lane-contiguous.
// lanes 0-15: N=lane, K=kc*32+j ; lanes 16-31: N=lane-16, K=kc*32+16+j
__global__ void pack_weights_kernel(const float* __restrict__ w,
                                    unsigned short* __restrict__ dst,
                                    int N, int Ktot, int taps) {
  int total = N * Ktot;
  for (int t = blockIdx.x * blockDim.x + threadIdx.x; t < total;
       t += gridDim.x * blockDim.x) {
    int j    = t & 15;
    int lane = (t >> 4) & 31;
    int rest = t >> 9;
    int nfcnt = N >> 4;
    int nf = rest % nfcnt;
    int kc = rest / nfcnt;
    int oc  = nf * 16 + (lane & 15);
    int K   = kc * 32 + ((lane >> 4) << 4) + j;
    int tap = K >> 6;
    int ic  = K & 63;
    dst[t] = f2bf(w[(oc * 64 + ic) * taps + tap]);
  }
}

// ---------------------- fused implicit-GEMM conv kernel ----------------------
// 256 threads = 8 waves. Workgroup: 8x8 pixel tile of one image.
// Weights are async-DMA staged into a double-buffered LDS region and shared by
// all 8 waves; WMMA overlaps the DMA of the next K-chunk.
// EPI: 0 = bias->store bf16 (RMB entry)      [NF=4,  TAPS=1]
//      1 = MU gating vs center h, store bf16 [NF=16, TAPS=9]
//      2 = bias + skip(x) in-place bf16      [NF=4,  TAPS=1]
//      3 = bias + Gumbel argmax sample f32   [NF=16, TAPS=1]
template <int NF, int TAPS, int EPI>
__global__ __launch_bounds__(256)
void wmma_conv_kernel(const unsigned short* __restrict__ act,
                      const unsigned short* __restrict__ wpk,
                      const float* __restrict__ bias,
                      unsigned short* __restrict__ out_bf,
                      float* __restrict__ out_f) {
  constexpr int NOUT = NF * 16;
  constexpr int HALO = (TAPS == 9) ? 1 : 0;
  constexpr int TP   = 8 + 2 * HALO;        // staged tile side
  constexpr int KCH  = TAPS * 2;            // K chunks of 32 (IC=64)
  constexpr int NFPW = NF / 2;              // N fragments per wave
  constexpr unsigned CHB = (unsigned)NF * 1024u;  // bytes of weights per chunk

  __shared__ unsigned short s_in[TP * TP * 64];
  __shared__ unsigned short s_w[2][NF * 512];     // double-buffered B chunks
  __shared__ float          s_g[64 * NOUT];

  const int b    = blockIdx.x;
  const int n    = b >> 8;
  const int tile = b & 255;
  const int y0   = (tile >> 4) << 3;
  const int x0   = (tile & 15) << 3;

  const unsigned sw_off[2] = { (unsigned)(uintptr_t)&s_w[0][0],
                               (unsigned)(uintptr_t)&s_w[1][0] };

  // ---- prologue: async-stage weight chunk 0 into buffer 0 ------------------
  for (unsigned o = threadIdx.x * 16u; o < CHB; o += 4096u)
    async_copy16(sw_off[0] + o, wpk, o);

  // ---- stage activation tile (+halo, zero-padded) into LDS -----------------
  for (int t = threadIdx.x; t < TP * TP * 8; t += 256) {
    int pix = t >> 3, ch = (t & 7) << 3;
    int yy = pix / TP, xx = pix - yy * TP;
    int gy = y0 - HALO + yy, gx = x0 - HALO + xx;
    uint4 v = {0u, 0u, 0u, 0u};
    if (gy >= 0 && gy < 128 && gx >= 0 && gx < 128)
      v = *(const uint4*)&act[(((size_t)n * 128 + gy) * 128 + gx) * 64 + ch];
    *(uint4*)&s_in[pix * 64 + ch] = v;
  }
  wait_async0();
  __syncthreads();

  const int wave  = threadIdx.x >> 5;
  const int lane  = threadIdx.x & 31;
  const int mfrag = wave & 3;
  const int nhalf = wave >> 2;
  const int p     = mfrag * 16 + (lane & 15);   // pixel in 8x8 tile
  const int ly    = p >> 3, lx = p & 7;
  const int sub   = (lane & 16) ? 16 : 0;       // A-layout K sub-base

  v8f acc[NFPW];
#pragma unroll
  for (int f = 0; f < NFPW; ++f)
#pragma unroll
    for (int r = 0; r < 8; ++r) acc[f][r] = 0.f;

  for (int kc = 0; kc < KCH; ++kc) {
    // issue async DMA of the next weight chunk into the other buffer
    if (kc + 1 < KCH) {
      unsigned nxt = sw_off[(kc + 1) & 1];
      for (unsigned o = threadIdx.x * 16u; o < CHB; o += 4096u)
        async_copy16(nxt + o, wpk, (unsigned)(kc + 1) * CHB + o);
    }

    const int tap = kc >> 1;
    const int icb = (kc & 1) << 5;
    const int ky = tap / 3, kx = tap - 3 * ky;  // 0 when TAPS==1
    const int spix = (TAPS == 9) ? ((ly + ky) * TP + lx + kx) : p;

    Frag16 a;
    a.q[0] = *(const uint4*)&s_in[spix * 64 + icb + sub];
    a.q[1] = *(const uint4*)&s_in[spix * 64 + icb + sub + 8];

    const unsigned short* swp = s_w[kc & 1];
#pragma unroll
    for (int f = 0; f < NFPW; ++f) {
      const unsigned short* bp =
          swp + ((unsigned)(nhalf * NFPW + f) * 32u + (unsigned)lane) * 16u;
      Frag16 bf;
      bf.q[0] = *(const uint4*)bp;
      bf.q[1] = *(const uint4*)(bp + 8);
      acc[f] = __builtin_amdgcn_wmma_f32_16x16x32_bf16(
          false, a.v, false, bf.v, (short)0, acc[f], false, false);
    }

    wait_async0();      // next chunk's DMA landed
    __syncthreads();    // all waves done reading current chunk
  }

  // ---- dump fragments into LDS result tile (C layout: lane<16 -> M=r) ------
#pragma unroll
  for (int f = 0; f < NFPW; ++f) {
    int nf   = nhalf * NFPW + f;
    int ncol = nf * 16 + (lane & 15);
#pragma unroll
    for (int r = 0; r < 8; ++r) {
      int m = (lane & 16) ? (r + 8) : r;
      s_g[(mfrag * 16 + m) * NOUT + ncol] = acc[f][r];
    }
  }
  __syncthreads();

  // ---- epilogues -----------------------------------------------------------
  if (EPI == 0) {  // entry conv: bias, store bf16 h
    for (int t = threadIdx.x; t < 64 * 64; t += 256) {
      int pp = t >> 6, c = t & 63;
      int lyy = pp >> 3, lxx = pp & 7;
      float v = s_g[pp * NOUT + c] + bias[c];
      out_bf[(((size_t)n * 128 + y0 + lyy) * 128 + x0 + lxx) * 64 + c] = f2bf(v);
    }
  } else if (EPI == 1) {  // MU: sigmoid/tanh gating with center h from s_in
    for (int t = threadIdx.x; t < 64 * 64; t += 256) {
      int pp = t >> 6, c = t & 63;
      int lyy = pp >> 3, lxx = pp & 7;
      float g1 = s_g[pp * NOUT + c]       + bias[c];
      float g2 = s_g[pp * NOUT + 64 + c]  + bias[64 + c];
      float g3 = s_g[pp * NOUT + 128 + c] + bias[128 + c];
      float uu = s_g[pp * NOUT + 192 + c] + bias[192 + c];
      float hc = bf2f(s_in[((lyy + HALO) * TP + lxx + HALO) * 64 + c]);
      float nh = sigf(g1) * tanhf(sigf(g2) * hc + sigf(g3) * tanhf(uu));
      out_bf[(((size_t)n * 128 + y0 + lyy) * 128 + x0 + lxx) * 64 + c] = f2bf(nh);
    }
  } else if (EPI == 2) {  // exit conv with residual skip, in-place on x buffer
    for (int t = threadIdx.x; t < 64 * 64; t += 256) {
      int pp = t >> 6, c = t & 63;
      int lyy = pp >> 3, lxx = pp & 7;
      size_t gi = (((size_t)n * 128 + y0 + lyy) * 128 + x0 + lxx) * 64 + c;
      float v = s_g[pp * NOUT + c] + bias[c] + bf2f(out_bf[gi]);
      out_bf[gi] = f2bf(v);
    }
  } else {  // logits + per-pixel Gumbel-max categorical sample
    if (threadIdx.x < 64) {
      int pp = threadIdx.x;
      int lyy = pp >> 3, lxx = pp & 7;
      int gy = y0 + lyy, gx = x0 + lxx;
      unsigned pix = ((unsigned)n << 14) | ((unsigned)gy << 7) | (unsigned)gx;
      float best = -3.4e38f; int bi = 0;
      for (int c = 0; c < 256; ++c) {
        float lg = s_g[pp * NOUT + c] + bias[c];
        unsigned s = pix * 2654435761u ^ ((unsigned)c * 0x9E3779B9u) ^ 0xA511E9B3u;
        s ^= s >> 16; s *= 0x85ebca6bu; s ^= s >> 13; s *= 0xc2b2ae35u; s ^= s >> 16;
        float u01 = ((s >> 8) + 1u) * (1.0f / 16777218.0f);
        float score = lg - __logf(-__logf(u01));
        if (score > best) { best = score; bi = c; }
      }
      out_f[(size_t)n * 16384 + (size_t)gy * 128 + gx] = (float)bi;
    }
  }
}

// ----------------------------------------------------------------------------
extern "C" void kernel_launch(void* const* d_in, const int* in_sizes, int n_in,
                              void* d_out, int out_size, void* d_ws, size_t ws_size,
                              hipStream_t stream) {
  (void)in_sizes; (void)n_in; (void)out_size; (void)ws_size;
  const float* inp = (const float*)d_in[0];

  // pytree (sorted dict key) order per RMB:
  // b_in, b_out, mu1.b, mu1.w, mu2.b, mu2.w, w_in, w_out
  const float *b_in[4], *b_out[4], *m1b[4], *m1w[4], *m2b[4], *m2w[4],
              *w_in[4], *w_out[4];
  for (int i = 0; i < 4; ++i) {
    int base = 1 + i * 8;
    b_in[i]  = (const float*)d_in[base + 0];
    b_out[i] = (const float*)d_in[base + 1];
    m1b[i]   = (const float*)d_in[base + 2];
    m1w[i]   = (const float*)d_in[base + 3];
    m2b[i]   = (const float*)d_in[base + 4];
    m2w[i]   = (const float*)d_in[base + 5];
    w_in[i]  = (const float*)d_in[base + 6];
    w_out[i] = (const float*)d_in[base + 7];
  }

  const size_t P = (size_t)16 * 128 * 128;
  unsigned short* xbuf = (unsigned short*)d_ws;   // NHWC bf16, 64 ch
  unsigned short* hA   = xbuf + P * 64;
  unsigned short* hB   = hA + P * 64;
  unsigned short* wsp  = hB + P * 64;
  unsigned short *pk_win[4], *pk_m1[4], *pk_m2[4], *pk_wout[4];
  for (int i = 0; i < 4; ++i) {
    pk_win[i]  = wsp; wsp += 64 * 64;
    pk_m1[i]   = wsp; wsp += 576 * 256;
    pk_m2[i]   = wsp; wsp += 576 * 256;
    pk_wout[i] = wsp; wsp += 64 * ((i < 3) ? 64 : 256);
  }

  dim3 blk(256);
  convert_input_kernel<<<2048, blk, 0, stream>>>(inp, xbuf);
  for (int i = 0; i < 4; ++i) {
    pack_weights_kernel<<<64,  blk, 0, stream>>>(w_in[i],  pk_win[i],  64,  64,  1);
    pack_weights_kernel<<<512, blk, 0, stream>>>(m1w[i],   pk_m1[i],   256, 576, 9);
    pack_weights_kernel<<<512, blk, 0, stream>>>(m2w[i],   pk_m2[i],   256, 576, 9);
    pack_weights_kernel<<<64,  blk, 0, stream>>>(w_out[i], pk_wout[i],
                                                 (i < 3) ? 64 : 256, 64, 1);
  }

  float* outf = (float*)d_out;
  for (int i = 0; i < 4; ++i) {
    wmma_conv_kernel<4, 1, 0><<<4096, blk, 0, stream>>>(xbuf, pk_win[i], b_in[i], hA, nullptr);
    wmma_conv_kernel<16, 9, 1><<<4096, blk, 0, stream>>>(hA, pk_m1[i], m1b[i], hB, nullptr);
    wmma_conv_kernel<16, 9, 1><<<4096, blk, 0, stream>>>(hB, pk_m2[i], m2b[i], hA, nullptr);
    if (i < 3)
      wmma_conv_kernel<4, 1, 2><<<4096, blk, 0, stream>>>(hA, pk_wout[i], b_out[i], xbuf, nullptr);
    else
      wmma_conv_kernel<16, 1, 3><<<4096, blk, 0, stream>>>(hA, pk_wout[i], b_out[i], nullptr, outf);
  }
}
